// ScannedRNN_4870492914074
// MI455X (gfx1250) — matrix operand order repaired and checked by reading, not compiled, not run on hardware
//
#include <hip/hip_runtime.h>

// ---------------------------------------------------------------------------
// GRU scan for MI455X (gfx1250, wave32, WMMA 16x16x32 bf16, fp32 accum).
//
//  Phase 0: pack Wi/Wh into WMMA B-fragment layout as bf16 hi/lo pairs (6 MB ws).
//  Phase 1: xp = x @ Wi + bi; each wave computes 8 N-tiles per A-fragment load.
//  Phase 2: sequential scan; 8 workgroups (16 batch rows each); h kept in LDS
//           as bf16 hi/lo in A-fragment element order (split once per step by
//           the writer, consumed via ds_load_b128 by all 32 waves); 9 WMMA per
//           K-chunk per column block (hi*hi + lo*hi + hi*lo -> ~2^-16 rel err).
//           Reset masking is folded into the h-write of the previous step.
//
//  Workspace layout (bytes):
//    [0,1.5M) WiH  [1.5M,3M) WiL  [3M,4.5M) WhH  [4.5M,6M) WhL
//    [6M, 6M+384MiB) xp tiles (T*8*96 tiles of 32 lanes x 8 f32, C-frag layout)
// ---------------------------------------------------------------------------

typedef __attribute__((ext_vector_type(16))) __bf16 v16bf;
typedef __attribute__((ext_vector_type(8)))  float  v8f;

namespace {
constexpr int Td = 512, Bd = 128, Hd = 512, N3 = 1536;
constexpr int KCH = 16;                   // K chunks of 32 over K=512
constexpr int NBLK = N3 / 16;             // 96 column blocks
constexpr int FRAG_U16 = 512;             // 32 lanes * 16 bf16 per fragment
constexpr int FRAGS_PER_MAT = NBLK * KCH; // 1536
}

union BF16x16 { v16bf v; unsigned short s[16]; uint4 q[2]; };

__device__ __forceinline__ unsigned short f2bf_rne(float f) {
  unsigned int u = __float_as_uint(f);
  u += 0x7FFFu + ((u >> 16) & 1u);
  return (unsigned short)(u >> 16);
}
__device__ __forceinline__ float bf2f(unsigned short s) {
  return __uint_as_float(((unsigned int)s) << 16);
}
// Truncation split: hi = top 16 bits (RTZ bf16), lo corrects it exactly to
// ~2^-16 relative. Much cheaper than RNE (no rounding-bit arithmetic).
__device__ __forceinline__ void split_trunc(float f, unsigned short& hi, unsigned short& lo) {
  unsigned int u = __float_as_uint(f);
  hi = (unsigned short)(u >> 16);
  float r = f - __uint_as_float(u & 0xffff0000u);
  lo = (unsigned short)(__float_as_uint(r) >> 16);
}
__device__ __forceinline__ v8f wmma_bf16(v16bf a, v16bf b, v8f c) {
  return __builtin_amdgcn_wmma_f32_16x16x32_bf16(false, a, false, b,
                                                 (short)0, c, false, false);
}
// Build A-fragment hi/lo from 16 fp32 values already in ISA lane-element order.
__device__ __forceinline__ void build_a_frag(const float* va, BF16x16& ah, BF16x16& al) {
#pragma unroll
  for (int e = 0; e < 16; ++e) split_trunc(va[e], ah.s[e], al.s[e]);
}

// ---------------------------------------------------------------------------
// Phase 0: pack W (512 x 1536, row-major) into B-fragment hi/lo arrays.
// B-frag (nb,ch): lane L holds col nb*16+(L&15), K = ch*32+(L>>4)*8+e+(e>>3)*8.
// ---------------------------------------------------------------------------
__global__ __launch_bounds__(256) void gru_pack_w(
    const float* __restrict__ Wi, const float* __restrict__ Wh,
    unsigned short* __restrict__ WiH, unsigned short* __restrict__ WiL,
    unsigned short* __restrict__ WhH, unsigned short* __restrict__ WhL) {
  const int lane = threadIdx.x & 31;
  const int wave = threadIdx.x >> 5;
  int f = blockIdx.x * 8 + wave;                    // 0..3071
  const float* W; unsigned short* Hh; unsigned short* Ll;
  if (f < FRAGS_PER_MAT) { W = Wi; Hh = WiH; Ll = WiL; }
  else                   { W = Wh; Hh = WhH; Ll = WhL; f -= FRAGS_PER_MAT; }
  const int nb = f >> 4, ch = f & 15;
  const int col = nb * 16 + (lane & 15);
  const int kb  = ch * 32 + (lane >> 4) * 8;
  const int off = f * FRAG_U16 + lane * 16;
#pragma unroll
  for (int e = 0; e < 16; ++e) {
    float w = W[(kb + e + (e >> 3) * 8) * N3 + col];
    unsigned short hi = f2bf_rne(w);          // one-time cost: keep RNE quality
    Hh[off + e] = hi;
    Ll[off + e] = f2bf_rne(w - bf2f(hi));
  }
}

// ---------------------------------------------------------------------------
// Phase 1: wave computes 8 tiles (same 16 rows, 8 adjacent N-blocks), so one
// A-fragment load/convert feeds 24 WMMAs. Group id = mt*12 + nbg.
// ---------------------------------------------------------------------------
__global__ __launch_bounds__(256) void gru_xproj(
    const float* __restrict__ x, const float* __restrict__ bi,
    const unsigned short* __restrict__ WiH, const unsigned short* __restrict__ WiL,
    float* __restrict__ xp) {
  const int lane = threadIdx.x & 31;
  const int wave = threadIdx.x >> 5;
  const int g   = blockIdx.x * 8 + wave;      // 0..49151
  const int nbg = g % 12;                     // N-group of 8 blocks
  const int mt  = g / 12;                     // 0..4095 = t*8 + mb
  const int kh  = lane >> 4;
  const int n   = lane & 15;
  const float* xrow = x + (size_t)(mt * 16 + n) * Hd;

  v8f acc[8];
#pragma unroll
  for (int j = 0; j < 8; ++j) acc[j] = (v8f){0.f,0.f,0.f,0.f,0.f,0.f,0.f,0.f};

  for (int ch = 0; ch < KCH; ++ch) {
    const float* p = xrow + ch * 32 + kh * 8;
    float4 a0 = *(const float4*)(p);
    float4 a1 = *(const float4*)(p + 4);
    float4 a2 = *(const float4*)(p + 16);
    float4 a3 = *(const float4*)(p + 20);
    float va[16] = {a0.x, a0.y, a0.z, a0.w, a1.x, a1.y, a1.z, a1.w,
                    a2.x, a2.y, a2.z, a2.w, a3.x, a3.y, a3.z, a3.w};
    BF16x16 ah, al;
    build_a_frag(va, ah, al);
#pragma unroll
    for (int j = 0; j < 8; ++j) {
      const int fo = ((nbg * 8 + j) * 16 + ch) * FRAG_U16 + lane * 16;
      v16bf bh = *(const v16bf*)(WiH + fo);
      v16bf bl = *(const v16bf*)(WiL + fo);
      acc[j] = wmma_bf16(ah.v, bh, acc[j]);
      acc[j] = wmma_bf16(al.v, bh, acc[j]);
      acc[j] = wmma_bf16(ah.v, bl, acc[j]);
    }
  }
#pragma unroll
  for (int j = 0; j < 8; ++j) {
    const int nb = nbg * 8 + j;
    const float bb = bi[nb * 16 + n];
    float* op = xp + (((size_t)mt * NBLK + nb) * 32 + lane) * 8;
    *(float4*)op       = make_float4(acc[j][0]+bb, acc[j][1]+bb, acc[j][2]+bb, acc[j][3]+bb);
    *(float4*)(op + 4) = make_float4(acc[j][4]+bb, acc[j][5]+bb, acc[j][6]+bb, acc[j][7]+bb);
  }
}

// ---------------------------------------------------------------------------
// Phase 2: sequential scan. blockIdx.x = batch block (16 rows); 32 waves, wave
// w owns output columns [16w,16w+16) via hp tiles nb = w, w+32, w+64 (r/z/n).
// h lives in LDS as bf16 hi/lo, element order == A-fragment order, so readers
// assemble fragments with two b128 LDS loads (no conversion in the K-loop).
// ---------------------------------------------------------------------------
__global__ __launch_bounds__(1024) void gru_scan(
    const unsigned char* __restrict__ resets, const float* __restrict__ h0,
    const unsigned short* __restrict__ WhH, const unsigned short* __restrict__ WhL,
    const float* __restrict__ bhn, const float* __restrict__ xp,
    float* __restrict__ out) {
  constexpr int HP = 520;                    // padded row stride (u16)
  __shared__ unsigned short hHi[16][HP];
  __shared__ unsigned short hLo[16][HP];
  const int tid  = threadIdx.x;
  const int lane = tid & 31;
  const int w    = tid >> 5;                 // wave id = column block
  const int mb   = blockIdx.x;               // batch block
  const int n    = lane & 15;
  const int kh   = lane >> 4;
  const int rm   = n;                        // A-fragment row for this lane

  // init: h0 with resets[0] mask applied, split into hi/lo
#pragma unroll
  for (int i = 0; i < 8; ++i) {
    int idx = tid + i * 1024;
    int r = idx >> 9, c = idx & 511;
    float v = resets[mb * 16 + r] ? 0.f : h0[(mb * 16 + r) * Hd + c];
    split_trunc(v, hHi[r][c], hLo[r][c]);
  }
  __syncthreads();

  const float bn = bhn[w * 16 + n];

  for (int t = 0; t < Td; ++t) {
    // issue xp loads early so they overlap the WMMA stream
    const size_t tb = (size_t)(t * 8 + mb) * NBLK;
    const float* pr = xp + ((tb + w     ) * 32 + lane) * 8;
    const float* pz = xp + ((tb + w + 32) * 32 + lane) * 8;
    const float* pn = xp + ((tb + w + 64) * 32 + lane) * 8;
    float4 r0 = *(const float4*)pr, r1 = *(const float4*)(pr + 4);
    float4 z0 = *(const float4*)pz, z1 = *(const float4*)(pz + 4);
    float4 n0 = *(const float4*)pn, n1 = *(const float4*)(pn + 4);
    // next-step reset flags for the 8 rows this lane will write (m = kh*8+v)
    uint2 rq = make_uint2(0u, 0u);
    if (t + 1 < Td)
      rq = *(const uint2*)(resets + (size_t)(t + 1) * Bd + mb * 16 + kh * 8);
    const unsigned char* rb = (const unsigned char*)&rq;

    v8f aR = {0.f,0.f,0.f,0.f,0.f,0.f,0.f,0.f};
    v8f aZ = aR, aN = aR;
    for (int ch = 0; ch < KCH; ++ch) {
      // A-fragment: two b128 LDS loads per half (hi, lo) — no VALU conversion
      const unsigned short* paH = &hHi[rm][ch * 32 + kh * 8];
      const unsigned short* paL = &hLo[rm][ch * 32 + kh * 8];
      BF16x16 ah, al;
      ah.q[0] = *(const uint4*)(paH);
      ah.q[1] = *(const uint4*)(paH + 16);
      al.q[0] = *(const uint4*)(paL);
      al.q[1] = *(const uint4*)(paL + 16);
      const int fb = ch * FRAG_U16 + lane * 16;
      v16bf bh, bl;
      bh = *(const v16bf*)(WhH + (w      ) * (16 * FRAG_U16) + fb);
      bl = *(const v16bf*)(WhL + (w      ) * (16 * FRAG_U16) + fb);
      aR = wmma_bf16(ah.v, bh, aR); aR = wmma_bf16(al.v, bh, aR); aR = wmma_bf16(ah.v, bl, aR);
      bh = *(const v16bf*)(WhH + (w + 32) * (16 * FRAG_U16) + fb);
      bl = *(const v16bf*)(WhL + (w + 32) * (16 * FRAG_U16) + fb);
      aZ = wmma_bf16(ah.v, bh, aZ); aZ = wmma_bf16(al.v, bh, aZ); aZ = wmma_bf16(ah.v, bl, aZ);
      bh = *(const v16bf*)(WhH + (w + 64) * (16 * FRAG_U16) + fb);
      bl = *(const v16bf*)(WhL + (w + 64) * (16 * FRAG_U16) + fb);
      aN = wmma_bf16(ah.v, bh, aN); aN = wmma_bf16(al.v, bh, aN); aN = wmma_bf16(ah.v, bl, aN);
    }

    float xr[8] = {r0.x, r0.y, r0.z, r0.w, r1.x, r1.y, r1.z, r1.w};
    float xz[8] = {z0.x, z0.y, z0.z, z0.w, z1.x, z1.y, z1.z, z1.w};
    float xn[8] = {n0.x, n0.y, n0.z, n0.w, n1.x, n1.y, n1.z, n1.w};

    float hnew[8];
#pragma unroll
    for (int v = 0; v < 8; ++v) {
      const int m = v + 8 * kh;               // C-layout row of this element
      const int col = w * 16 + n;
      const float hp = bf2f(hHi[m][col]) + bf2f(hLo[m][col]);
      const float rg = 1.f / (1.f + expf(-(xr[v] + aR[v])));
      const float zg = 1.f / (1.f + expf(-(xz[v] + aZ[v])));
      const float ng = tanhf(xn[v] + rg * (aN[v] + bn));
      const float hv = (1.f - zg) * ng + zg * hp;
      hnew[v] = hv;
      out[((size_t)t * Bd + mb * 16 + m) * Hd + col] = hv;
    }
    __syncthreads();                          // all LDS reads of h done
#pragma unroll
    for (int v = 0; v < 8; ++v) {             // write h for t+1 (mask folded in)
      const int m = v + 8 * kh;
      const float hw = rb[v] ? 0.f : hnew[v];
      split_trunc(hw, hHi[m][w * 16 + n], hLo[m][w * 16 + n]);
    }
    __syncthreads();                          // h ready for next step
  }
}

// ---------------------------------------------------------------------------
extern "C" void kernel_launch(void* const* d_in, const int* in_sizes, int n_in,
                              void* d_out, int out_size, void* d_ws, size_t ws_size,
                              hipStream_t stream) {
  const float*         x      = (const float*)d_in[0];
  const unsigned char* resets = (const unsigned char*)d_in[1];  // jnp.bool_ -> 1 byte
  const float*         h0     = (const float*)d_in[2];
  const float*         Wi     = (const float*)d_in[3];
  const float*         bi     = (const float*)d_in[4];
  const float*         Wh     = (const float*)d_in[5];
  const float*         bhn    = (const float*)d_in[6];
  float* out = (float*)d_out;

  char* ws = (char*)d_ws;
  unsigned short* WiH = (unsigned short*)ws;
  unsigned short* WiL = WiH + (size_t)FRAGS_PER_MAT * FRAG_U16;
  unsigned short* WhH = WiL + (size_t)FRAGS_PER_MAT * FRAG_U16;
  unsigned short* WhL = WhH + (size_t)FRAGS_PER_MAT * FRAG_U16;
  float* xp = (float*)(ws + 4 * (size_t)FRAGS_PER_MAT * FRAG_U16 * 2);  // +6 MB

  // Phase 0: weight packing (3072 fragments, 8 per block)
  gru_pack_w<<<384, 256, 0, stream>>>(Wi, Wh, WiH, WiL, WhH, WhL);
  // Phase 1: input projection (4096 M-tiles x 12 N-groups, 8 groups per block)
  gru_xproj<<<6144, 256, 0, stream>>>(x, bi, WiH, WiL, xp);
  // Phase 2: sequential scan, one workgroup per 16 batch rows
  gru_scan<<<8, 1024, 0, stream>>>(resets, h0, WhH, WhL, bhn, xp, out);
}
